// DynamicRoutingMoEQuadraticNeuralNetworks_44659069944352
// MI455X (gfx1250) — matrix operand order repaired and checked by reading, compiled, not run on hardware
//
#include <hip/hip_runtime.h>
#include <cstdint>
#include <cstddef>

// ---------------------------------------------------------------------------
// Types for CDNA5 WMMA (wave32): v_wmma_f32_16x16x32_bf16
// ---------------------------------------------------------------------------
typedef __bf16 v16bf __attribute__((ext_vector_type(16)));
typedef float  v8f   __attribute__((ext_vector_type(8)));
typedef float         f32x4 __attribute__((ext_vector_type(4)));
typedef unsigned int  u32x4 __attribute__((ext_vector_type(4)));
typedef unsigned int  u32x2 __attribute__((ext_vector_type(2)));

enum Act { ACT_NONE = 0, ACT_RELU = 1, ACT_TANH = 2, ACT_ELU = 3 };

__device__ __forceinline__ unsigned int f32_to_bf16(float f) {
  unsigned int u = __float_as_uint(f);
  u += 0x7fffu + ((u >> 16) & 1u);      // round-to-nearest-even
  return u >> 16;
}
__device__ __forceinline__ unsigned int pack_bf16x2(float lo, float hi) {
  return f32_to_bf16(lo) | (f32_to_bf16(hi) << 16);
}

union FragAB { v16bf v; unsigned int u[8]; };
union FragC  { v8f   v; float f[8]; };

#define TM 128
#define TN 64
#define TK 32
#define LDS_STRIDE 40   // ushorts per row (32 data + 8 pad); 80B rows, 16B slots

// ---------------------------------------------------------------------------
// GEMM: C = act(A[M,K](bf16) @ W[N,K](bf16)^T + bias) (+resid) (scaled +=)
// Contract: K % 32 == 0, lda % 8 == 0, ldw % 8 == 0 (elements).
// 256 threads = 8 waves in a 4x2 grid; each wave owns a 32x32 patch
// (2x2 accumulators) -> 4 WMMAs per k-step with A/B fragment reuse.
// Double-buffered LDS; b128 global loads; zero conversion VALU in hot loop.
// ---------------------------------------------------------------------------
__global__ __launch_bounds__(256) void gemm_wmma_kernel(
    const unsigned short* __restrict__ A, long lda,
    const unsigned short* __restrict__ W, long ldw,
    const float* __restrict__ bias,
    float* __restrict__ Cf,               // fp32 output (or accumulate target)
    unsigned short* __restrict__ Cbf,     // bf16 output (optional)
    const float* __restrict__ resid,
    const float* __restrict__ rowscale,
    int rs_shift, int rs_stride,
    int M, int N, int K, int act)
{
  __shared__ __align__(16) unsigned short As[2][TM][LDS_STRIDE];
  __shared__ __align__(16) unsigned short Bs[2][TN][LDS_STRIDE];

  const int tid  = threadIdx.x;
  const int m0   = blockIdx.y * TM;
  const int n0   = blockIdx.x * TN;
  const int wave = tid >> 5;
  const int lane = tid & 31;
  const int wm   = wave >> 1;       // 0..3  (32-row band)
  const int wn   = wave & 1;        // 0..1  (32-col band)
  const int half = lane >> 4;
  const int r    = lane & 15;

  FragC acc[2][2];
#pragma unroll
  for (int i = 0; i < 2; ++i)
#pragma unroll
    for (int j = 0; j < 2; ++j)
#pragma unroll
      for (int e = 0; e < 8; ++e) acc[i][j].f[e] = 0.f;

  const int lrow = tid >> 2;        // 0..63
  const int lk   = (tid & 3) * 8;   // 0,8,16,24
  const int gmA0 = m0 + lrow;
  const int gmA1 = m0 + lrow + 64;
  const int gnB  = n0 + lrow;
  const bool am0 = (gmA0 < M);
  const bool am1 = (gmA1 < M);
  const bool bn  = (gnB < N);
  const unsigned short* __restrict__ arow0 = A + (long)gmA0 * lda + lk;
  const unsigned short* __restrict__ arow1 = A + (long)gmA1 * lda + lk;
  const unsigned short* __restrict__ wrow  = W + (long)gnB * ldw + lk;

  const int nk = K / TK;
  u32x4 ra0 = (u32x4)0u, ra1 = (u32x4)0u, rw = (u32x4)0u;

  // prologue: tile 0
  if (am0) ra0 = *(const u32x4*)arow0;
  if (am1) ra1 = *(const u32x4*)arow1;
  if (bn)  rw  = *(const u32x4*)wrow;
  *(u32x4*)&As[0][lrow][lk]      = ra0;
  *(u32x4*)&As[0][lrow + 64][lk] = ra1;
  *(u32x4*)&Bs[0][lrow][lk]      = rw;

  for (int kt = 0; kt < nk; ++kt) {
    const int cur = kt & 1;
    if (kt + 1 < nk) {                 // issue next tile's loads early
      const long ko = (long)(kt + 1) * TK;
      if (am0) ra0 = *(const u32x4*)(arow0 + ko);
      if (am1) ra1 = *(const u32x4*)(arow1 + ko);
      if (bn)  rw  = *(const u32x4*)(wrow + ko);
    }
    __syncthreads();                   // stage 'cur' visible

    FragAB fa[2], fb[2];
#pragma unroll
    for (int v = 0; v < 8; ++v) {
      const int ka = ((v & 3) * 2) + half * 8 + ((v >> 2) * 16); // A 16x32 layout
      const int kb = ((v & 3) * 2) + half * 8 + ((v >> 2) * 16); // B 32x16 layout
      fa[0].u[v] = *(const unsigned int*)&As[cur][wm * 32 + r][ka];
      fa[1].u[v] = *(const unsigned int*)&As[cur][wm * 32 + 16 + r][ka];
      fb[0].u[v] = *(const unsigned int*)&Bs[cur][wn * 32 + r][kb];
      fb[1].u[v] = *(const unsigned int*)&Bs[cur][wn * 32 + 16 + r][kb];
    }
#pragma unroll
    for (int i = 0; i < 2; ++i)
#pragma unroll
      for (int j = 0; j < 2; ++j)
        acc[i][j].v = __builtin_amdgcn_wmma_f32_16x16x32_bf16(
            false, fa[i].v, false, fb[j].v, (short)0, acc[i][j].v, false, false);
    __syncthreads();                   // reads of 'cur' complete

    if (kt + 1 < nk) {
      *(u32x4*)&As[cur ^ 1][lrow][lk]      = ra0;
      *(u32x4*)&As[cur ^ 1][lrow + 64][lk] = ra1;
      *(u32x4*)&Bs[cur ^ 1][lrow][lk]      = rw;
    }
  }

#pragma unroll
  for (int i = 0; i < 2; ++i) {
#pragma unroll
    for (int j = 0; j < 2; ++j) {
      const int mbase = m0 + wm * 32 + i * 16 + half * 8;
      const int ncol  = n0 + wn * 32 + j * 16 + r;
#pragma unroll
      for (int e = 0; e < 8; ++e) {
        const int mrow = mbase + e;
        if (mrow < M && ncol < N) {
          float val = acc[i][j].f[e];
          if (bias) val += bias[ncol];
          if (act == ACT_RELU)      val = fmaxf(val, 0.f);
          else if (act == ACT_TANH) val = tanhf(val);
          else if (act == ACT_ELU)  val = (val > 0.f) ? val : expm1f(val);
          if (resid) val += resid[(long)mrow * N + ncol];
          const long o = (long)mrow * N + ncol;
          if (rowscale) {
            Cf[o] += rowscale[(long)(mrow >> rs_shift) * rs_stride] * val;
          } else {
            if (Cf)  Cf[o]  = val;
            if (Cbf) Cbf[o] = (unsigned short)f32_to_bf16(val);
          }
        }
      }
    }
  }
}

// ---------------------------------------------------------------------------
// fp32 -> bf16 bulk convert (8 elems / thread)
// ---------------------------------------------------------------------------
__global__ void f2bf_kernel(const float* __restrict__ in,
                            unsigned short* __restrict__ out, long n8) {
  const long i = (long)blockIdx.x * blockDim.x + threadIdx.x;
  if (i >= n8) return;
  const f32x4 a = ((const f32x4*)in)[2 * i];
  const f32x4 b = ((const f32x4*)in)[2 * i + 1];
  u32x4 u = { pack_bf16x2(a[0], a[1]), pack_bf16x2(a[2], a[3]),
              pack_bf16x2(b[0], b[1]), pack_bf16x2(b[2], b[3]) };
  ((u32x4*)out)[i] = u;
}

// ---------------------------------------------------------------------------
// Block reductions (blockDim == 256)
// ---------------------------------------------------------------------------
__device__ __forceinline__ float blk_sum(float v, float* red) {
  const int t = threadIdx.x;
  red[t] = v; __syncthreads();
  for (int s = 128; s > 0; s >>= 1) { if (t < s) red[t] += red[t + s]; __syncthreads(); }
  const float r = red[0]; __syncthreads(); return r;
}
__device__ __forceinline__ float blk_max(float v, float* red) {
  const int t = threadIdx.x;
  red[t] = v; __syncthreads();
  for (int s = 128; s > 0; s >>= 1) { if (t < s) red[t] = fmaxf(red[t], red[t + s]); __syncthreads(); }
  const float r = red[0]; __syncthreads(); return r;
}

// ---------------------------------------------------------------------------
// Per-row feature stats + sorted-|x| concentration + fused k-predictor MLP
// ---------------------------------------------------------------------------
__global__ __launch_bounds__(256) void features_kernel(
    const float* __restrict__ x,
    const float* __restrict__ kw1, const float* __restrict__ kb1,
    const float* __restrict__ kw2, const float* __restrict__ kb2,
    float* __restrict__ feats, float* __restrict__ kvals)
{
  __shared__ float s[1024];
  __shared__ float red[256];
  const int b = blockIdx.x, t = threadIdx.x;
  const float* xr = x + (size_t)b * 1024;
  float zc = 0.f, sm = 0.f, ss = 0.f, mx = 0.f, sa = 0.f;
  for (int i = t; i < 1024; i += 256) {
    const float v = xr[i];
    const float a = fabsf(v);
    s[i] = a;
    if (v == 0.f) zc += 1.f;
    sm += v; ss += v * v; sa += a;
    mx = fmaxf(mx, a);
  }
  const float ZC = blk_sum(zc, red);
  const float SM = blk_sum(sm, red);
  const float SS = blk_sum(ss, red);
  const float SA = blk_sum(sa, red);
  const float MX = blk_max(mx, red);
  const float mean = SM * (1.f / 1024.f);
  const float var  = (SS - 1024.f * mean * mean) * (1.f / 1023.f);
  const float stdv = sqrtf(var + 1e-8f);
  float sk = 0.f;
  for (int i = t; i < 1024; i += 256) {
    const float z = (xr[i] - mean) / stdv;
    sk += z * z * z;
  }
  const float SK = blk_sum(sk, red) * (1.f / 1024.f);
  for (unsigned ksz = 2; ksz <= 1024; ksz <<= 1) {
    for (unsigned j = ksz >> 1; j > 0; j >>= 1) {
      __syncthreads();
      for (unsigned i = (unsigned)t; i < 1024; i += 256) {
        const unsigned l = i ^ j;
        if (l > i) {
          const bool up = ((i & ksz) == 0);
          const float a = s[i], c = s[l];
          if ((a > c) == up) { s[i] = c; s[l] = a; }
        }
      }
    }
  }
  __syncthreads();
  float tp = 0.f;
  for (int i = 820 + t; i < 1024; i += 256) tp += s[i];   // top 204 = int(0.2*1024)
  const float TP = blk_sum(tp, red);
  if (t == 0) {
    float fv[6];
    fv[0] = ZC * (1.f / 1024.f);
    fv[1] = var;
    fv[2] = MX;
    fv[3] = sqrtf(SS);
    fv[4] = SK;
    fv[5] = TP / (SA + 1e-8f);
    for (int i = 0; i < 6; ++i) feats[b * 6 + i] = fv[i];
    float o = kb2[0];
    for (int j = 0; j < 16; ++j) {
      float h = kb1[j];
      for (int i = 0; i < 6; ++i) h += kw1[j * 6 + i] * fv[i];
      h = fmaxf(h, 0.f);
      o += kw2[j] * h;
    }
    kvals[b] = 1.f + 3.f / (1.f + expf(-o));
  }
}

// ---------------------------------------------------------------------------
// Exact median of 8192 k_vals (single workgroup, LDS bitonic), floor+clip
// ---------------------------------------------------------------------------
__global__ __launch_bounds__(1024) void median_kernel(const float* __restrict__ kv,
                                                      int* __restrict__ kout)
{
  __shared__ float s[8192];
  const int t = threadIdx.x;
  for (int i = t; i < 8192; i += 1024) s[i] = kv[i];
  for (unsigned ksz = 2; ksz <= 8192; ksz <<= 1) {
    for (unsigned j = ksz >> 1; j > 0; j >>= 1) {
      __syncthreads();
      for (unsigned i = (unsigned)t; i < 8192; i += 1024) {
        const unsigned l = i ^ j;
        if (l > i) {
          const bool up = ((i & ksz) == 0);
          const float a = s[i], c = s[l];
          if ((a > c) == up) { s[i] = c; s[l] = a; }
        }
      }
    }
  }
  __syncthreads();
  if (t == 0) {
    const float med = 0.5f * (s[4095] + s[4096]);
    int k = (int)floorf(med);
    if (k < 1) k = 1;
    if (k > 4) k = 4;
    *kout = k;
  }
}

// ---------------------------------------------------------------------------
// spec = sigmoid(esp_out + sim_out); xsb = concat(xbf, spec, pad) (bf16, 1056)
// ---------------------------------------------------------------------------
#define XSLD 1056
__global__ void spec_xs_kernel(const unsigned short* __restrict__ xbf,
                               const float* __restrict__ espo,
                               const float* __restrict__ simo,
                               unsigned short* __restrict__ xsb)
{
  const int b = blockIdx.x, t = threadIdx.x;
  unsigned short* xr = xsb + (size_t)b * XSLD;
  const unsigned short* xin = xbf + (size_t)b * 1024;
  for (int i = t; i < 1024; i += 256) xr[i] = xin[i];
  if (t < 32) {
    unsigned short v = 0;
    if (t < 6) {
      const float z = espo[b * 6 + t] + simo[b * 6 + t];
      v = (unsigned short)f32_to_bf16(1.f / (1.f + expf(-z)));
    }
    xr[1024 + t] = v;
  }
}

// pack rout_w1 (512 x 1030 fp32) into zero-padded bf16 (512 x 1056)
__global__ void pack_w_kernel(const float* __restrict__ w, unsigned short* __restrict__ wp) {
  const long i = (long)blockIdx.x * blockDim.x + threadIdx.x;
  if (i >= (long)512 * XSLD) return;
  const long n = i / XSLD, k = i % XSLD;
  wp[i] = (k < 1030) ? (unsigned short)f32_to_bf16(w[n * 1030 + k]) : (unsigned short)0;
}

// ---------------------------------------------------------------------------
// softmax(logits) -> top4 -> gate-softmax over first k -> per-expert coeffs
// ---------------------------------------------------------------------------
__global__ void gates_kernel(const float* __restrict__ logits,
                             const int* __restrict__ kptr,
                             float* __restrict__ gcoef, int Brows)
{
  const int b = blockIdx.x * blockDim.x + threadIdx.x;
  if (b >= Brows) return;
  float p[6];
  float m = -1e30f;
  for (int e = 0; e < 6; ++e) { p[e] = logits[b * 6 + e]; m = fmaxf(m, p[e]); }
  float den = 0.f;
  for (int e = 0; e < 6; ++e) { p[e] = expf(p[e] - m); den += p[e]; }
  for (int e = 0; e < 6; ++e) p[e] /= den;
  int taken = 0; float tp[4]; int ti[4];
  for (int s = 0; s < 4; ++s) {
    float best = -1.f; int bi = 0;
    for (int e = 0; e < 6; ++e)
      if (!((taken >> e) & 1) && p[e] > best) { best = p[e]; bi = e; }
    tp[s] = best; ti[s] = bi; taken |= 1 << bi;
  }
  const int k = *kptr;
  const float mm = tp[0];
  float d2 = 0.f, g[4] = {0.f, 0.f, 0.f, 0.f};
  for (int s = 0; s < k; ++s) { g[s] = expf(tp[s] - mm); d2 += g[s]; }
  float out[6] = {0.f, 0.f, 0.f, 0.f, 0.f, 0.f};
  for (int s = 0; s < k; ++s) out[ti[s]] += g[s] / d2;
  for (int e = 0; e < 6; ++e) gcoef[b * 6 + e] = out[e];
}

// ---------------------------------------------------------------------------
// CrossField expert: per-row qkv + 32x32 field attention in LDS -> bf16 out
// ---------------------------------------------------------------------------
__global__ __launch_bounds__(256) void cf_attn_kernel(
    const float* __restrict__ x, const float* __restrict__ w_in,
    const float* __restrict__ b_in, unsigned short* __restrict__ o)
{
  __shared__ float xf[32][32];
  __shared__ float Wl[96][32];
  __shared__ float Q[32][33];
  __shared__ float Kf[32][33];
  __shared__ float Vf[32][33];
  __shared__ float S[32][33];
  const int b = blockIdx.x, t = threadIdx.x;
  const float* xr = x + (size_t)b * 1024;
  for (int i = t; i < 1024; i += 256) xf[i >> 5][i & 31] = xr[i];
  for (int i = t; i < 96 * 32; i += 256) Wl[i >> 5][i & 31] = w_in[i];
  __syncthreads();
  for (int idx = t; idx < 32 * 96; idx += 256) {
    const int f = idx / 96, c = idx % 96;
    float acc = b_in[c];
    for (int i = 0; i < 32; ++i) acc += xf[f][i] * Wl[c][i];
    if (c < 32) Q[f][c] = acc;
    else if (c < 64) Kf[f][c - 32] = acc;
    else Vf[f][c - 64] = acc;
  }
  __syncthreads();
  for (int idx = t; idx < 1024; idx += 256) {
    const int i = idx >> 5, j = idx & 31;
    float d = 0.f;
    for (int q = 0; q < 32; ++q) d += Q[i][q] * Kf[j][q];
    S[i][j] = d * 0.17677669529663687f;     // 1/sqrt(32)
  }
  __syncthreads();
  if (t < 32) {
    float m = -1e30f;
    for (int j = 0; j < 32; ++j) m = fmaxf(m, S[t][j]);
    float den = 0.f;
    for (int j = 0; j < 32; ++j) { const float e = expf(S[t][j] - m); S[t][j] = e; den += e; }
    const float inv = 1.f / den;
    for (int j = 0; j < 32; ++j) S[t][j] *= inv;
  }
  __syncthreads();
  unsigned short* orow = o + (size_t)b * 1024;
  for (int idx = t; idx < 1024; idx += 256) {
    const int i = idx >> 5, j = idx & 31;
    float acc = 0.f;
    for (int q = 0; q < 32; ++q) acc += S[i][q] * Vf[q][j];
    orow[i * 32 + j] = (unsigned short)f32_to_bf16(acc);
  }
}

// ---------------------------------------------------------------------------
// Elementwise / combine kernels
// ---------------------------------------------------------------------------
__global__ void mul_bf_kernel(const float* __restrict__ a, const float* __restrict__ b,
                              unsigned short* __restrict__ o, long n8) {
  const long i = (long)blockIdx.x * blockDim.x + threadIdx.x;
  if (i >= n8) return;
  const f32x4 a0 = ((const f32x4*)a)[2 * i], a1 = ((const f32x4*)a)[2 * i + 1];
  const f32x4 b0 = ((const f32x4*)b)[2 * i], b1 = ((const f32x4*)b)[2 * i + 1];
  const f32x4 p0 = a0 * b0, p1 = a1 * b1;
  u32x4 u = { pack_bf16x2(p0[0], p0[1]), pack_bf16x2(p0[2], p0[3]),
              pack_bf16x2(p1[0], p1[1]), pack_bf16x2(p1[2], p1[3]) };
  ((u32x4*)o)[i] = u;
}
__global__ void de_combine_kernel(const float* __restrict__ x, const float* __restrict__ mlp,
                                  const float* __restrict__ gatel, const float* __restrict__ g,
                                  float* __restrict__ comb, long n4) {
  const long i = (long)blockIdx.x * blockDim.x + threadIdx.x;
  if (i >= n4) return;
  const int b = (int)(i >> 8);
  const float gg = g[b * 6 + 1];
  const f32x4 xv = ((const f32x4*)x)[i];
  const f32x4 mv = ((const f32x4*)mlp)[i];
  const f32x4 gl = ((const f32x4*)gatel)[i];
  f32x4 rr = ((f32x4*)comb)[i];
#pragma unroll
  for (int c = 0; c < 4; ++c) {
    const float sg = 1.f / (1.f + expf(-gl[c]));
    rr[c] += gg * (mv[c] + sg * xv[c]);
  }
  ((f32x4*)comb)[i] = rr;
}
__global__ void hf_combine_kernel(const float* __restrict__ x, const float* __restrict__ filt,
                                  const float* __restrict__ g, float* __restrict__ comb, long n4) {
  const long i = (long)blockIdx.x * blockDim.x + threadIdx.x;
  if (i >= n4) return;
  const int b = (int)(i >> 8);
  const float gg = g[b * 6 + 3];
  const f32x4 xv = ((const f32x4*)x)[i];
  const f32x4 fv = ((const f32x4*)filt)[i];
  f32x4 rr = ((f32x4*)comb)[i];
#pragma unroll
  for (int c = 0; c < 4; ++c) rr[c] += gg * (xv[c] + (fv[c] - xv[c]) * xv[c]);
  ((f32x4*)comb)[i] = rr;
}
// last combiner: writes fp32 comb AND its bf16 copy for the final GEMM
__global__ void lt_combine_kernel(const float* __restrict__ x, const float* __restrict__ tv,
                                  const float* __restrict__ g, float* __restrict__ comb,
                                  unsigned short* __restrict__ combbf, long n4) {
  const long i = (long)blockIdx.x * blockDim.x + threadIdx.x;
  if (i >= n4) return;
  const int b = (int)(i >> 8);
  const float gg = g[b * 6 + 5];
  const f32x4 xv = ((const f32x4*)x)[i];
  const f32x4 t4 = ((const f32x4*)tv)[i];
  f32x4 rr = ((f32x4*)comb)[i];
#pragma unroll
  for (int c = 0; c < 4; ++c) {
    const float sg = (xv[c] > 0.f) ? 1.f : ((xv[c] < 0.f) ? -1.f : 0.f);
    rr[c] += gg * sg * sqrtf(fabsf(t4[c] * xv[c]) + 1e-8f);
  }
  ((f32x4*)comb)[i] = rr;
  u32x2 u = { pack_bf16x2(rr[0], rr[1]), pack_bf16x2(rr[2], rr[3]) };
  ((u32x2*)combbf)[i] = u;
}
__global__ void tp_kernel(const float* __restrict__ x,
                          const float* __restrict__ tcw, const float* __restrict__ tcb,
                          const float* __restrict__ tfw, const float* __restrict__ tfb,
                          const float* __restrict__ g, float* __restrict__ comb, long n) {
  const long i = (long)blockIdx.x * blockDim.x + threadIdx.x;
  if (i >= n) return;
  const int b = (int)(i >> 10), pos = (int)(i & 1023);
  const float xc = x[i];
  const float xm = (pos > 0) ? x[i - 1] : 0.f;
  const float xp = (pos < 1023) ? x[i + 1] : 0.f;
  float a = tfb[0];
#pragma unroll
  for (int c = 0; c < 4; ++c) {
    float cv = tcw[c * 3 + 0] * xm + tcw[c * 3 + 1] * xc + tcw[c * 3 + 2] * xp + tcb[c];
    cv = fmaxf(cv, 0.f);
    a += tfw[c] * cv;
  }
  const float wt = 1.f / (1.f + expf(-a));
  comb[i] += g[b * 6 + 4] * xc * wt * xc;
}

// ---------------------------------------------------------------------------
// Host-side launch
// ---------------------------------------------------------------------------
typedef unsigned short bfp;

static inline void gemm(hipStream_t s,
                        const bfp* A, long lda, const bfp* W, long ldw,
                        const float* bias, float* Cf, bfp* Cbf,
                        const float* resid, const float* rowscale,
                        int rs_shift, int rs_stride, int M, int N, int K, int act)
{
  dim3 grid((unsigned)((N + TN - 1) / TN), (unsigned)((M + TM - 1) / TM));
  gemm_wmma_kernel<<<grid, dim3(256), 0, s>>>(A, lda, W, ldw, bias, Cf, Cbf, resid,
                                              rowscale, rs_shift, rs_stride, M, N, K, act);
}
static inline void f2bf(hipStream_t s, const float* in, bfp* out, long n) {
  const long n8 = n / 8;
  f2bf_kernel<<<(unsigned)((n8 + 255) / 256), 256, 0, s>>>(in, out, n8);
}

extern "C" void kernel_launch(void* const* d_in, const int* in_sizes, int n_in,
                              void* d_out, int out_size, void* d_ws, size_t ws_size,
                              hipStream_t stream)
{
  (void)in_sizes; (void)n_in; (void)out_size; (void)ws_size;
  const int B = 8192, D = 1024;
  const long BD = (long)B * D;

  const float* x       = (const float*)d_in[0];
  const float* esp     = (const float*)d_in[1];
  const float* sim_w1  = (const float*)d_in[2];
  const float* sim_b1  = (const float*)d_in[3];
  const float* sim_w2  = (const float*)d_in[4];
  const float* sim_b2  = (const float*)d_in[5];
  const float* rout_w1 = (const float*)d_in[6];
  const float* rout_b1 = (const float*)d_in[7];
  const float* rout_w2 = (const float*)d_in[8];
  const float* rout_b2 = (const float*)d_in[9];
  const float* kp_w1   = (const float*)d_in[10];
  const float* kp_b1   = (const float*)d_in[11];
  const float* kp_w2   = (const float*)d_in[12];
  const float* kp_b2   = (const float*)d_in[13];
  const float* sp_v_w  = (const float*)d_in[14];
  const float* sp_v_b  = (const float*)d_in[15];
  const float* sp_o_w  = (const float*)d_in[16];
  const float* sp_o_b  = (const float*)d_in[17];
  const float* sp_q1_w = (const float*)d_in[18];
  const float* sp_q1_b = (const float*)d_in[19];
  const float* sp_q2_w = (const float*)d_in[20];
  const float* sp_q2_b = (const float*)d_in[21];
  const float* de_w1   = (const float*)d_in[22];
  const float* de_b1   = (const float*)d_in[23];
  const float* de_w2   = (const float*)d_in[24];
  const float* de_b2   = (const float*)d_in[25];
  const float* de_g_w  = (const float*)d_in[26];
  const float* de_g_b  = (const float*)d_in[27];
  const float* cf_in_w = (const float*)d_in[28];
  const float* cf_in_b = (const float*)d_in[29];
  const float* cf_o_w  = (const float*)d_in[30];
  const float* cf_o_b  = (const float*)d_in[31];
  const float* cf_f_w  = (const float*)d_in[32];
  const float* cf_f_b  = (const float*)d_in[33];
  const float* hf_w1   = (const float*)d_in[34];
  const float* hf_b1   = (const float*)d_in[35];
  const float* hf_w2   = (const float*)d_in[36];
  const float* hf_b2   = (const float*)d_in[37];
  const float* tc_w    = (const float*)d_in[38];
  const float* tc_b    = (const float*)d_in[39];
  const float* tf_w    = (const float*)d_in[40];
  const float* tf_b    = (const float*)d_in[41];
  const float* lt_w1   = (const float*)d_in[42];
  const float* lt_b1   = (const float*)d_in[43];
  const float* lt_w2   = (const float*)d_in[44];
  const float* lt_b2   = (const float*)d_in[45];
  const float* op_w    = (const float*)d_in[46];
  const float* op_b    = (const float*)d_in[47];

  // ---- fp32 workspace ----
  float* comb  = (float*)d_ws;
  float* t2    = comb + BD;
  float* t3    = t2 + BD;
  float* s_sim = t3 + BD;
  float* s_esp = s_sim + (size_t)B * 6;
  float* feats = s_esp + (size_t)B * 6;
  float* kvals = feats + (size_t)B * 6;
  float* gcoef = kvals + B;
  int*   kint  = (int*)(gcoef + (size_t)B * 6);
  // ---- bf16 workspace (16B aligned: preceding float count is mult of 4) ----
  bfp* xbf  = (bfp*)(kint + 16);
  bfp* t1b  = xbf + BD;
  bfp* t3b  = t1b + BD;
  bfp* xsb  = t3b + BD;                      // B x 1056
  bfp* hb   = xsb + (size_t)B * XSLD;        // B x 512 hidden (also B x 256)
  bfp* wb   = hb + (size_t)B * 512;
  bfp* w_sim1 = wb;
  bfp* w_sim2 = w_sim1 + 512 * 1024;
  bfp* w_esp  = w_sim2 + 6 * 512;
  bfp* w_r1p  = w_esp + 6 * 1024;
  bfp* w_r2   = w_r1p + 512 * XSLD;
  bfp* w_spv  = w_r2 + 6 * 512;
  bfp* w_spo  = w_spv + 1024 * 1024;
  bfp* w_spq1 = w_spo + 1024 * 1024;
  bfp* w_spq2 = w_spq1 + 1024 * 1024;
  bfp* w_de1  = w_spq2 + 1024 * 1024;
  bfp* w_de2  = w_de1 + 256 * 1024;
  bfp* w_deg  = w_de2 + 1024 * 256;
  bfp* w_cfo  = w_deg + 1024 * 1024;
  bfp* w_cff  = w_cfo + 32 * 32;
  bfp* w_hf1  = w_cff + 32 * 32;
  bfp* w_hf2  = w_hf1 + 256 * 1024;
  bfp* w_lt1  = w_hf2 + 1024 * 256;
  bfp* w_lt2  = w_lt1 + 256 * 1024;
  bfp* w_op   = w_lt2 + 1024 * 256;

  const unsigned EW4 = (unsigned)(BD / 4 / 256);
  const unsigned EW8 = (unsigned)(BD / 8 / 256);
  const unsigned EWS = (unsigned)((BD + 255) / 256);

  // ---- one-time conversions (weights + input) ----
  f2bf(stream, x, xbf, BD);
  f2bf(stream, sim_w1, w_sim1, 512 * 1024);
  f2bf(stream, sim_w2, w_sim2, 6 * 512);
  f2bf(stream, esp, w_esp, 6 * 1024);
  f2bf(stream, rout_w2, w_r2, 6 * 512);
  f2bf(stream, sp_v_w, w_spv, 1024 * 1024);
  f2bf(stream, sp_o_w, w_spo, 1024 * 1024);
  f2bf(stream, sp_q1_w, w_spq1, 1024 * 1024);
  f2bf(stream, sp_q2_w, w_spq2, 1024 * 1024);
  f2bf(stream, de_w1, w_de1, 256 * 1024);
  f2bf(stream, de_w2, w_de2, 1024 * 256);
  f2bf(stream, de_g_w, w_deg, 1024 * 1024);
  f2bf(stream, cf_o_w, w_cfo, 32 * 32);
  f2bf(stream, cf_f_w, w_cff, 32 * 32);
  f2bf(stream, hf_w1, w_hf1, 256 * 1024);
  f2bf(stream, hf_w2, w_hf2, 1024 * 256);
  f2bf(stream, lt_w1, w_lt1, 256 * 1024);
  f2bf(stream, lt_w2, w_lt2, 1024 * 256);
  f2bf(stream, op_w, w_op, 1024 * 1024);
  pack_w_kernel<<<(512 * XSLD + 255) / 256, 256, 0, stream>>>(rout_w1, w_r1p);

  // ---- Router ----
  gemm(stream, xbf, D, w_sim1, 1024, sim_b1, nullptr, hb, nullptr, nullptr, 0, 0, B, 512, 1024, ACT_RELU);
  gemm(stream, hb, 512, w_sim2, 512, sim_b2, s_sim, nullptr, nullptr, nullptr, 0, 0, B, 6, 512, ACT_NONE);
  gemm(stream, xbf, D, w_esp, 1024, nullptr, s_esp, nullptr, nullptr, nullptr, 0, 0, B, 6, 1024, ACT_NONE);
  spec_xs_kernel<<<B, 256, 0, stream>>>(xbf, s_esp, s_sim, xsb);
  gemm(stream, xsb, XSLD, w_r1p, XSLD, rout_b1, nullptr, hb, nullptr, nullptr, 0, 0, B, 512, XSLD, ACT_RELU);
  gemm(stream, hb, 512, w_r2, 512, rout_b2, s_sim, nullptr, nullptr, nullptr, 0, 0, B, 6, 512, ACT_NONE);
  features_kernel<<<B, 256, 0, stream>>>(x, kp_w1, kp_b1, kp_w2, kp_b2, feats, kvals);
  median_kernel<<<1, 1024, 0, stream>>>(kvals, kint);
  gates_kernel<<<(B + 255) / 256, 256, 0, stream>>>(s_sim, kint, gcoef, B);
  hipMemsetAsync(comb, 0, (size_t)BD * sizeof(float), stream);

  // ---- SparseQuadratic expert (gcoef[...,0]) ----
  gemm(stream, xbf, D, w_spv, 1024, sp_v_b, nullptr, t1b, nullptr, nullptr, 0, 0, B, 1024, 1024, ACT_NONE);
  gemm(stream, t1b, D, w_spo, 1024, sp_o_b, t2, nullptr, nullptr, nullptr, 0, 0, B, 1024, 1024, ACT_NONE);
  mul_bf_kernel<<<EW8, 256, 0, stream>>>(t2, x, t3b, BD / 8);
  gemm(stream, t3b, D, w_spq1, 1024, sp_q1_b, nullptr, t1b, nullptr, nullptr, 0, 0, B, 1024, 1024, ACT_RELU);
  gemm(stream, t1b, D, w_spq2, 1024, sp_q2_b, comb, nullptr, nullptr, gcoef + 0, 0, 6, B, 1024, 1024, ACT_NONE);

  // ---- DenseQuadratic expert (gcoef[...,1]) ----
  gemm(stream, xbf, D, w_de1, 1024, de_b1, nullptr, hb, nullptr, nullptr, 0, 0, B, 256, 1024, ACT_RELU);
  gemm(stream, hb, 256, w_de2, 256, de_b2, t2, nullptr, nullptr, nullptr, 0, 0, B, 1024, 256, ACT_NONE);
  gemm(stream, xbf, D, w_deg, 1024, de_g_b, t3, nullptr, nullptr, nullptr, 0, 0, B, 1024, 1024, ACT_NONE);
  de_combine_kernel<<<EW4, 256, 0, stream>>>(x, t2, t3, gcoef, comb, BD / 4);

  // ---- CrossFieldQuadratic expert (gcoef[...,2]) ----
  cf_attn_kernel<<<B, 256, 0, stream>>>(x, cf_in_w, cf_in_b, t1b);
  gemm(stream, t1b, 32, w_cfo, 32, cf_o_b, t2, nullptr, nullptr, nullptr, 0, 0, B * 32, 32, 32, ACT_NONE);
  mul_bf_kernel<<<EW8, 256, 0, stream>>>(t2, x, t3b, BD / 8);
  gemm(stream, t3b, 32, w_cff, 32, cf_f_b, comb, nullptr, nullptr, gcoef + 2, 5, 6, B * 32, 32, 32, ACT_NONE);

  // ---- HighFreqQuadratic expert (gcoef[...,3]) ----
  gemm(stream, xbf, D, w_hf1, 1024, hf_b1, nullptr, hb, nullptr, nullptr, 0, 0, B, 256, 1024, ACT_TANH);
  gemm(stream, hb, 256, w_hf2, 256, hf_b2, t2, nullptr, nullptr, nullptr, 0, 0, B, 1024, 256, ACT_NONE);
  hf_combine_kernel<<<EW4, 256, 0, stream>>>(x, t2, gcoef, comb, BD / 4);

  // ---- TemporalQuadratic expert (gcoef[...,4]) ----
  tp_kernel<<<EWS, 256, 0, stream>>>(x, tc_w, tc_b, tf_w, tf_b, gcoef, comb, BD);

  // ---- LongTailQuadratic expert (gcoef[...,5]) ---- (emits comb fp32 + bf16)
  gemm(stream, xbf, D, w_lt1, 1024, lt_b1, nullptr, hb, nullptr, nullptr, 0, 0, B, 256, 1024, ACT_ELU);
  gemm(stream, hb, 256, w_lt2, 256, lt_b2, t2, nullptr, nullptr, nullptr, 0, 0, B, 1024, 256, ACT_NONE);
  lt_combine_kernel<<<EW4, 256, 0, stream>>>(x, t2, gcoef, comb, t1b, BD / 4);

  // ---- Output projection + residual ----
  gemm(stream, t1b, D, w_op, 1024, op_b, (float*)d_out, nullptr, x, nullptr, 0, 0, B, 1024, 1024, ACT_NONE);
}
